// HybridTextClassifier_65481071404683
// MI455X (gfx1250) — matrix operand order repaired and checked by reading, not compiled, not run on hardware
//
#include <hip/hip_runtime.h>
#include <hip/hip_bf16.h>
#include <stdint.h>

// ---------------- problem constants ----------------
#define BB   16
#define HH   64
#define WW   64
#define EE   512
#define FFN  2048
#define LL   4
#define NHD  8
#define PP   1024            // (64/2)*(64/2)
#define DKH  64              // EE / NHD
#define MM   (BB * PP)       // 16384 token rows

typedef unsigned short us;
typedef __bf16 v16bf __attribute__((ext_vector_type(16)));
typedef float  v8f   __attribute__((ext_vector_type(8)));
typedef int    v4i_n __attribute__((vector_size(16)));   // native 4xi32 vector

union Frag {
    int4  i2[2];
    v16bf v;
};

#if defined(__HIP_DEVICE_COMPILE__) &&                                   \
    __has_builtin(__builtin_amdgcn_global_load_async_to_lds_b128) &&     \
    __has_builtin(__builtin_amdgcn_s_wait_asynccnt)
#define USE_ASYNC_LDS 1
#else
#define USE_ASYNC_LDS 0
#endif

// copy 16 bytes global -> LDS (async on CDNA5 if available)
static __device__ __forceinline__ void cp_lds_b128(us* dst, const us* src) {
#if USE_ASYNC_LDS
    typedef __attribute__((address_space(1))) v4i_n gv4i;   // global
    typedef __attribute__((address_space(3))) v4i_n lv4i;   // LDS
    __builtin_amdgcn_global_load_async_to_lds_b128(
        (gv4i*)(us*)src, (lv4i*)dst, 0, 0);
#else
    *(int4*)dst = *(const int4*)src;
#endif
}
static __device__ __forceinline__ void cp_lds_wait() {
#if USE_ASYNC_LDS
    __builtin_amdgcn_s_wait_asynccnt(0);
#endif
}

static __device__ __forceinline__ us f2bf(float f) {
    unsigned u = __float_as_uint(f);
    unsigned r = (u + 0x7FFFu + ((u >> 16) & 1u)) >> 16;
    return (us)r;
}

static __device__ __forceinline__ v8f wmma_bf16(const Frag& a, const Frag& b, v8f c) {
    return __builtin_amdgcn_wmma_f32_16x16x32_bf16(false, a.v, false, b.v, (short)0, c,
                                                   false, false);
}

// reduce across each 16-lane half (rows of a WMMA C tile)
static __device__ __forceinline__ float hmax16(float v) {
    #pragma unroll
    for (int m = 8; m >= 1; m >>= 1) v = fmaxf(v, __shfl_xor(v, m, 32));
    return v;
}
static __device__ __forceinline__ float hsum16(float v) {
    #pragma unroll
    for (int m = 8; m >= 1; m >>= 1) v += __shfl_xor(v, m, 32);
    return v;
}

// full 256-thread block sum (8 waves)
static __device__ __forceinline__ float block_sum_256(float v) {
    __shared__ float red[8];
    const int lane = threadIdx.x & 31, wid = threadIdx.x >> 5;
    #pragma unroll
    for (int m = 16; m >= 1; m >>= 1) v += __shfl_xor(v, m, 32);
    if (lane == 0) red[wid] = v;
    __syncthreads();
    float t = 0.f;
    #pragma unroll
    for (int i = 0; i < 8; i++) t += red[i];
    __syncthreads();
    return t;
}

// ---------------- fp32 [L][N][K] -> bf16 k-major [L][K][N] ----------------
__global__ void convert_t_f32_bf16(const float* __restrict__ src, us* __restrict__ dst,
                                   int Nd, int Kd, int total) {
    int idx = blockIdx.x * 256 + threadIdx.x;
    if (idx >= total) return;
    int nk = Nd * Kd;
    int l = idx / nk, r = idx - l * nk;
    int k = r / Nd, n = r - k * Nd;
    dst[(size_t)l * nk + (size_t)k * Nd + n] = f2bf(src[(size_t)l * nk + (size_t)n * Kd + k]);
}

// ---------------- patch conv + Linear(1->E) + positional encoding ----------------
__global__ __launch_bounds__(256)
void embed_kernel(const float* __restrict__ data, const float* __restrict__ conv_w,
                  const float* __restrict__ conv_b, const float* __restrict__ lin_w,
                  const float* __restrict__ lin_b, float* __restrict__ xf,
                  us* __restrict__ xb) {
    int idx = blockIdx.x * 256 + threadIdx.x;         // < B*P*E
    int e  = idx & (EE - 1);
    int bp = idx >> 9;
    int p  = bp & (PP - 1);
    int b  = bp >> 10;
    int ph = p >> 5, pw = p & 31;
    const float* dp = data + (size_t)b * (HH * WW) + (ph * 2) * WW + pw * 2;
    float conv = dp[0] * conv_w[0] + dp[1] * conv_w[1] +
                 dp[WW] * conv_w[2] + dp[WW + 1] * conv_w[3] + conv_b[0];
    int i2 = e & ~1;
    float freq = __expf(-(float)i2 * (9.210340371976184f / (float)EE)); // ln(10000)/E
    float ang  = (float)p * freq;
    float pe   = (e & 1) ? __cosf(ang) : __sinf(ang);
    float v = conv * lin_w[e] + lin_b[e] + pe;
    xf[idx] = v;
    xb[idx] = f2bf(v);
}

// ---------------- bf16 WMMA GEMM ----------------
// Y[M,N] = act(X[M,K] @ Wt[K,N] + bias); Wt is pre-transposed (k-major).
// BM=256, BN=64, BK=64; 8 waves, each wave owns 32 rows x 64 cols (8 C tiles).
// Double-buffered LDS with async global->LDS copies overlapping the WMMAs.
// kt_out!=0: bf16 output written in [b, h, d, p] (d-major) layout for attention K.
__global__ __launch_bounds__(256)
void gemm_bf16_kernel(const us* __restrict__ Xb, const us* __restrict__ Wt,
                      const float* __restrict__ bias, float* __restrict__ Yf,
                      us* __restrict__ Yb, int Mtot, int Ntot, int Ktot,
                      int relu, int kt_out) {
    __shared__ __align__(16) us Xs[2][256 * 64];   // 2 x 32 KB
    __shared__ __align__(16) us Ws[2][64 * 64];    // 2 x  8 KB
    const int tid  = threadIdx.x;
    const int lane = tid & 31;
    const int wm   = tid >> 5;          // wave id 0..7 -> 32-row stripe
    const int mlo  = lane & 15;
    const int half = lane >> 4;
    const int m0 = blockIdx.x * 256;
    const int n0 = blockIdx.y * 64;
    const int kd = mlo + 16 * half;

    auto load_tiles = [&](int k0, int buf) {
        // X tile: 256 rows x 64 bf16 = 2048 b128 chunks, 8 per thread
        #pragma unroll
        for (int it = 0; it < 8; it++) {
            int c = it * 256 + tid;
            int r = c >> 3, off = (c & 7) * 8;
            cp_lds_b128(&Xs[buf][r * 64 + off],
                        Xb + (size_t)(m0 + r) * Ktot + k0 + off);
        }
        // W tile (k-major): 64 rows x 64 bf16 = 512 chunks, 2 per thread
        #pragma unroll
        for (int it = 0; it < 2; it++) {
            int c = it * 256 + tid;
            int r = c >> 3, off = (c & 7) * 8;
            cp_lds_b128(&Ws[buf][r * 64 + off],
                        Wt + (size_t)(k0 + r) * Ntot + n0 + off);
        }
    };

    v8f acc[2][4];
    #pragma unroll
    for (int s = 0; s < 2; s++)
        #pragma unroll
        for (int t = 0; t < 4; t++)
            #pragma unroll
            for (int r = 0; r < 8; r++) acc[s][t][r] = 0.f;

    const int nTiles = Ktot >> 6;
    load_tiles(0, 0);
    cp_lds_wait();
    __syncthreads();

    for (int i = 0; i < nTiles; i++) {
        const int cur = i & 1;
        if (i + 1 < nTiles) load_tiles((i + 1) << 6, cur ^ 1);

        #pragma unroll
        for (int ks = 0; ks < 2; ks++) {
            Frag a[2];
            #pragma unroll
            for (int s = 0; s < 2; s++) {
                const us* ap = &Xs[cur][(wm * 32 + s * 16 + mlo) * 64 + ks * 32 + 8 * half];
                a[s].i2[0] = *(const int4*)ap;
                a[s].i2[1] = *(const int4*)(ap + 16);
            }
            #pragma unroll
            for (int t = 0; t < 4; t++) {
                Frag bf;
                const us* bp = &Ws[cur][(ks * 32 + 8 * half) * 64 + t * 16 + mlo];
                #pragma unroll
                for (int j = 0; j < 8; j++) {
                    ((us*)&bf)[j]     = bp[j * 64];
                    ((us*)&bf)[8 + j] = bp[(16 + j) * 64];
                }
                acc[0][t] = wmma_bf16(a[0], bf, acc[0][t]);
                acc[1][t] = wmma_bf16(a[1], bf, acc[1][t]);
            }
        }
        cp_lds_wait();
        __syncthreads();
    }

    #pragma unroll
    for (int s = 0; s < 2; s++)
        #pragma unroll
        for (int t = 0; t < 4; t++) {
            int col = n0 + t * 16 + mlo;
            float bv = bias[col];
            #pragma unroll
            for (int r = 0; r < 8; r++) {
                int row = m0 + wm * 32 + s * 16 + r + 8 * half;
                float v = acc[s][t][r] + bv;
                if (relu) v = fmaxf(v, 0.f);
                if (Yf) Yf[(size_t)row * Ntot + col] = v;
                if (Yb) {
                    if (kt_out) {   // [b][h][d][p] layout (Ntot must be EE)
                        int p  = row & (PP - 1);
                        int bi = row >> 10;
                        int hh = col >> 6;
                        int d  = col & 63;
                        Yb[(((size_t)(bi * NHD + hh)) * DKH + d) * PP + p] = f2bf(v);
                    } else {
                        Yb[(size_t)row * Ntot + col] = f2bf(v);
                    }
                }
            }
        }
}

// ---------------- flash attention: one (b,h), 64 query rows per block ----------------
// KtG layout: [B][NH][d=64][p=1024] (written by K-projection with kt_out)
__global__ __launch_bounds__(128)
void attn_kernel(const us* __restrict__ Qb, const us* __restrict__ KtG,
                 const us* __restrict__ Vb, us* __restrict__ Ab) {
    __shared__ __align__(16) us Kt[2][64 * 32];   // [d=64][key=32] double-buffered
    __shared__ __align__(16) us Vs[2][32 * 64];   // [key=32][d=64] double-buffered
    __shared__ __align__(16) us Ps[4 * 16 * 32];  // per-wave prob staging 16x32

    const int tid  = threadIdx.x;
    const int lane = tid & 31;
    const int wm   = tid >> 5;
    const int mlo  = lane & 15;
    const int half = lane >> 4;
    const int b = blockIdx.x >> 3;
    const int h = blockIdx.x & 7;
    const int q0 = blockIdx.y * 64 + wm * 16;
    const us* ktBase = KtG + (size_t)(b * NHD + h) * DKH * PP;

    auto load_kv = [&](int kb0, int buf) {
        // K tile (d-major source): thread -> (drow, 16-key half)
        int drow = tid >> 1, koff = (tid & 1) * 16;
        const us* kg = ktBase + (size_t)drow * PP + kb0 + koff;
        cp_lds_b128(&Kt[buf][drow * 32 + koff], kg);
        cp_lds_b128(&Kt[buf][drow * 32 + koff + 8], kg + 8);
        // V tile (natural): thread -> (key row, 16-d chunk)
        int kc = tid >> 2, d0 = (tid & 3) * 16;
        const us* vg = Vb + (size_t)(b * PP + kb0 + kc) * EE + h * DKH + d0;
        cp_lds_b128(&Vs[buf][kc * 64 + d0], vg);
        cp_lds_b128(&Vs[buf][kc * 64 + d0 + 8], vg + 8);
    };

    // Q A-fragments for d = 0..31 and 32..63
    size_t qoff = (size_t)(b * PP + q0 + mlo) * EE + h * DKH;
    Frag a0, a1;
    a0.i2[0] = *(const int4*)(Qb + qoff + 8 * half);
    a0.i2[1] = *(const int4*)(Qb + qoff + 16 + 8 * half);
    a1.i2[0] = *(const int4*)(Qb + qoff + 32 + 8 * half);
    a1.i2[1] = *(const int4*)(Qb + qoff + 48 + 8 * half);

    float mrow[8], lrow[8];
    v8f o[4];
    #pragma unroll
    for (int r = 0; r < 8; r++) { mrow[r] = -1e30f; lrow[r] = 0.f; }
    #pragma unroll
    for (int t = 0; t < 4; t++)
        #pragma unroll
        for (int r = 0; r < 8; r++) o[t][r] = 0.f;

    const int kd = mlo + 16 * half;
    const int nBlk = PP / 32;

    load_kv(0, 0);
    cp_lds_wait();
    __syncthreads();

    for (int i = 0; i < nBlk; i++) {
        const int cur = i & 1;
        if (i + 1 < nBlk) load_kv((i + 1) * 32, cur ^ 1);

        // scores tiles: s0 = keys 0..15, s1 = keys 16..31 (contract over d=64)
        v8f s0, s1;
        #pragma unroll
        for (int r = 0; r < 8; r++) { s0[r] = 0.f; s1[r] = 0.f; }
        {
            Frag bk;
            const int4* p;
            (void)p;
            const us* kp = &Kt[cur][(8 * half) * 32 + mlo];
            #pragma unroll
            for (int j = 0; j < 8; j++) { ((us*)&bk)[j] = kp[j * 32]; ((us*)&bk)[8 + j] = kp[(16 + j) * 32]; }
            s0 = wmma_bf16(a0, bk, s0);
            #pragma unroll
            for (int j = 0; j < 8; j++) { ((us*)&bk)[j] = kp[(32 + j) * 32]; ((us*)&bk)[8 + j] = kp[(48 + j) * 32]; }
            s0 = wmma_bf16(a1, bk, s0);
            #pragma unroll
            for (int j = 0; j < 8; j++) { ((us*)&bk)[j] = kp[j * 32 + 16]; ((us*)&bk)[8 + j] = kp[(16 + j) * 32 + 16]; }
            s1 = wmma_bf16(a0, bk, s1);
            #pragma unroll
            for (int j = 0; j < 8; j++) { ((us*)&bk)[j] = kp[(32 + j) * 32 + 16]; ((us*)&bk)[8 + j] = kp[(48 + j) * 32 + 16]; }
            s1 = wmma_bf16(a1, bk, s1);
        }

        // streaming softmax in C layout: VGPR r -> row (r + 8*half), cols across 16 lanes
        float alpha[8];
        #pragma unroll
        for (int r = 0; r < 8; r++) {
            float v0 = s0[r] * 0.125f;   // 1/sqrt(64)
            float v1 = s1[r] * 0.125f;
            float rmax = hmax16(fmaxf(v0, v1));
            float mnew = fmaxf(mrow[r], rmax);
            float al   = __expf(mrow[r] - mnew);
            float p0   = __expf(v0 - mnew);
            float p1   = __expf(v1 - mnew);
            lrow[r] = lrow[r] * al + hsum16(p0 + p1);
            mrow[r] = mnew;
            alpha[r] = al;
            int Mr = r + 8 * half;
            Ps[wm * 512 + Mr * 32 + mlo]      = f2bf(p0);
            Ps[wm * 512 + Mr * 32 + 16 + mlo] = f2bf(p1);
        }
        #pragma unroll
        for (int t = 0; t < 4; t++)
            #pragma unroll
            for (int r = 0; r < 8; r++) o[t][r] *= alpha[r];

        // reload probabilities as A fragment (same-wave LDS RAW, in-order)
        Frag pa;
        pa.i2[0] = *(const int4*)(Ps + wm * 512 + mlo * 32 + 8 * half);
        pa.i2[1] = *(const int4*)(Ps + wm * 512 + mlo * 32 + 16 + 8 * half);

        #pragma unroll
        for (int t = 0; t < 4; t++) {
            Frag bv;
            const us* vp = &Vs[cur][(8 * half) * 64 + t * 16 + mlo];
            #pragma unroll
            for (int j = 0; j < 8; j++) {
                ((us*)&bv)[j]     = vp[j * 64];
                ((us*)&bv)[8 + j] = vp[(16 + j) * 64];
            }
            o[t] = wmma_bf16(pa, bv, o[t]);
        }
        cp_lds_wait();
        __syncthreads();
    }

    #pragma unroll
    for (int t = 0; t < 4; t++)
        #pragma unroll
        for (int r = 0; r < 8; r++) {
            float v = o[t][r] / lrow[r];
            int row = b * PP + blockIdx.y * 64 + wm * 16 + r + 8 * half;
            int col = h * DKH + t * 16 + mlo;
            Ab[(size_t)row * EE + col] = f2bf(v);
        }
}

// ---------------- residual add + LayerNorm ----------------
__global__ __launch_bounds__(256)
void add_ln_kernel(float* __restrict__ xf, us* __restrict__ xb,
                   const float* __restrict__ proj, const float* __restrict__ g,
                   const float* __restrict__ bb) {
    const int row = blockIdx.x, tid = threadIdx.x;
    size_t base = (size_t)row * EE;
    float s0 = xf[base + tid]       + proj[base + tid];
    float s1 = xf[base + tid + 256] + proj[base + tid + 256];
    float mu = block_sum_256(s0 + s1) * (1.0f / EE);
    float d0 = s0 - mu, d1 = s1 - mu;
    float var = block_sum_256(d0 * d0 + d1 * d1) * (1.0f / EE);
    float inv = rsqrtf(var + 1e-5f);
    float y0 = d0 * inv * g[tid]       + bb[tid];
    float y1 = d1 * inv * g[tid + 256] + bb[tid + 256];
    xf[base + tid]       = y0;
    xf[base + tid + 256] = y1;
    xb[base + tid]       = f2bf(y0);
    xb[base + tid + 256] = f2bf(y1);
}

// ---------------- mean-pool over P + classifier ----------------
__global__ __launch_bounds__(256)
void pool_cls_kernel(const float* __restrict__ xf, const float* __restrict__ Wc,
                     const float* __restrict__ bc, float* __restrict__ out) {
    const int b = blockIdx.x, tid = threadIdx.x;
    float acc = 0.f;
    for (int e = tid; e < EE; e += 256) {
        const float* px = xf + (size_t)b * PP * EE + e;
        float s = 0.f;
        for (int p = 0; p < PP; p++) s += px[(size_t)p * EE];
        acc += (s * (1.0f / PP)) * Wc[e];
    }
    float tot = block_sum_256(acc);
    if (tid == 0) out[b] = tot + bc[0];
}

// ---------------- host orchestration ----------------
extern "C" void kernel_launch(void* const* d_in, const int* in_sizes, int n_in,
                              void* d_out, int out_size, void* d_ws, size_t ws_size,
                              hipStream_t stream) {
    (void)in_sizes; (void)n_in; (void)out_size; (void)ws_size;
    const float* data   = (const float*)d_in[0];
    const float* conv_w = (const float*)d_in[1];
    const float* conv_b = (const float*)d_in[2];
    const float* lin_w  = (const float*)d_in[3];
    const float* lin_b  = (const float*)d_in[4];
    const float* Wq = (const float*)d_in[5];   const float* bq = (const float*)d_in[6];
    const float* Wk = (const float*)d_in[7];   const float* bk = (const float*)d_in[8];
    const float* Wv = (const float*)d_in[9];   const float* bv = (const float*)d_in[10];
    const float* Wo = (const float*)d_in[11];  const float* bo = (const float*)d_in[12];
    const float* ln1_g = (const float*)d_in[13]; const float* ln1_b = (const float*)d_in[14];
    const float* ln2_g = (const float*)d_in[15]; const float* ln2_b = (const float*)d_in[16];
    const float* W1 = (const float*)d_in[17];  const float* b1 = (const float*)d_in[18];
    const float* W2 = (const float*)d_in[19];  const float* b2 = (const float*)d_in[20];
    const float* Wc = (const float*)d_in[21];  const float* bc = (const float*)d_in[22];
    float* out = (float*)d_out;

    // workspace carving (256B aligned chunks)
    unsigned char* wp = (unsigned char*)d_ws;
    auto take = [&](size_t bytes) -> void* {
        void* p = wp;
        wp += (bytes + 255) & ~(size_t)255;
        return p;
    };
    float* xf   = (float*)take((size_t)MM * EE * 4);
    us*    xb   = (us*)take((size_t)MM * EE * 2);
    float* proj = (float*)take((size_t)MM * EE * 4);
    us*    un   = (us*)take((size_t)MM * FFN * 2);  // union: {q,k,v,attn-out} / {ffn hidden}
    us* qb = un;
    us* kb = un + (size_t)MM * EE;      // holds K in [b][h][d][p] layout
    us* vb = un + (size_t)2 * MM * EE;
    us* ab = un + (size_t)3 * MM * EE;
    us* h1 = un;
    us* wqb = (us*)take((size_t)LL * EE * EE * 2);
    us* wkb = (us*)take((size_t)LL * EE * EE * 2);
    us* wvb = (us*)take((size_t)LL * EE * EE * 2);
    us* wob = (us*)take((size_t)LL * EE * EE * 2);
    us* w1b = (us*)take((size_t)LL * FFN * EE * 2);
    us* w2b = (us*)take((size_t)LL * EE * FFN * 2);

    // convert + transpose weights to bf16 k-major [K][N]
    int nEE = LL * EE * EE, nFE = LL * FFN * EE;
    convert_t_f32_bf16<<<(nEE + 255) / 256, 256, 0, stream>>>(Wq, wqb, EE, EE, nEE);
    convert_t_f32_bf16<<<(nEE + 255) / 256, 256, 0, stream>>>(Wk, wkb, EE, EE, nEE);
    convert_t_f32_bf16<<<(nEE + 255) / 256, 256, 0, stream>>>(Wv, wvb, EE, EE, nEE);
    convert_t_f32_bf16<<<(nEE + 255) / 256, 256, 0, stream>>>(Wo, wob, EE, EE, nEE);
    convert_t_f32_bf16<<<(nFE + 255) / 256, 256, 0, stream>>>(W1, w1b, FFN, EE, nFE);
    convert_t_f32_bf16<<<(nFE + 255) / 256, 256, 0, stream>>>(W2, w2b, EE, FFN, nFE);

    // embedding
    embed_kernel<<<(size_t)MM * EE / 256, 256, 0, stream>>>(data, conv_w, conv_b,
                                                            lin_w, lin_b, xf, xb);

    dim3 gEE(MM / 256, EE / 64);
    dim3 gFF(MM / 256, FFN / 64);
    dim3 gAT(BB * NHD, PP / 64);

    for (int l = 0; l < LL; l++) {
        const us* wq = wqb + (size_t)l * EE * EE;
        const us* wk = wkb + (size_t)l * EE * EE;
        const us* wv = wvb + (size_t)l * EE * EE;
        const us* wo = wob + (size_t)l * EE * EE;
        const us* w1 = w1b + (size_t)l * FFN * EE;
        const us* w2 = w2b + (size_t)l * EE * FFN;

        gemm_bf16_kernel<<<gEE, 256, 0, stream>>>(xb, wq, bq + l * EE, nullptr, qb,
                                                  MM, EE, EE, 0, 0);
        gemm_bf16_kernel<<<gEE, 256, 0, stream>>>(xb, wk, bk + l * EE, nullptr, kb,
                                                  MM, EE, EE, 0, 1);   // d-major K
        gemm_bf16_kernel<<<gEE, 256, 0, stream>>>(xb, wv, bv + l * EE, nullptr, vb,
                                                  MM, EE, EE, 0, 0);
        attn_kernel<<<gAT, 128, 0, stream>>>(qb, kb, vb, ab);
        gemm_bf16_kernel<<<gEE, 256, 0, stream>>>(ab, wo, bo + l * EE, proj, nullptr,
                                                  MM, EE, EE, 0, 0);
        add_ln_kernel<<<MM, 256, 0, stream>>>(xf, xb, proj, ln1_g + l * EE, ln1_b + l * EE);
        gemm_bf16_kernel<<<gFF, 256, 0, stream>>>(xb, w1, b1 + l * FFN, nullptr, h1,
                                                  MM, FFN, EE, 1, 0);
        gemm_bf16_kernel<<<gEE, 256, 0, stream>>>(h1, w2, b2 + l * EE, proj, nullptr,
                                                  MM, EE, FFN, 0, 0);
        add_ln_kernel<<<MM, 256, 0, stream>>>(xf, xb, proj, ln2_g + l * EE, ln2_b + l * EE);
    }

    pool_cls_kernel<<<BB, 256, 0, stream>>>(xf, Wc, bc, out);
}
